// KnowledgeAugmentedGlobalAttention_2559800508843
// MI455X (gfx1250) — matrix operand order repaired and checked by reading, not compiled, not run on hardware
//
#include <hip/hip_runtime.h>
#include <hip/hip_bf16.h>

// ---------------------------------------------------------------------------
// KnowledgeAugmentedGlobalAttention for MI455X (gfx1250, wave32, WMMA)
// B=2, L=2048, E=1024, H=16, D=64
// ---------------------------------------------------------------------------

#define B_  2
#define L_  2048
#define E_  1024
#define H_  16
#define D_  64
#define EQ_ 3072   // 3*E, qkv row stride

typedef _Float16 v16h __attribute__((ext_vector_type(16)));
typedef _Float16 h8   __attribute__((ext_vector_type(8)));
typedef float    v8f  __attribute__((ext_vector_type(8)));
typedef float    f4   __attribute__((ext_vector_type(4)));

// Load a WMMA f16 fragment from two 16-byte chunks (p0[0..7] -> elems 0..7,
// p1[0..7] -> elems 8..15).  Each chunk is one global_load_b128 / ds_load_b128.
__device__ __forceinline__ v16h load_frag(const _Float16* p0, const _Float16* p1) {
    h8 lo = *(const h8*)p0;
    h8 hi = *(const h8*)p1;
    v16h r;
#pragma unroll
    for (int i = 0; i < 8; ++i) { r[i] = lo[i]; r[i + 8] = hi[i]; }
    return r;
}

__device__ __forceinline__ v8f wmma16(v16h a, v16h b, v8f c) {
    return __builtin_amdgcn_wmma_f32_16x16x32_f16(false, a, false, b,
                                                  (short)0, c, false, false);
}

// ---------------------------------------------------------------------------
// Kernel 1: f32 -> f16 conversion, 8 elements / thread
// ---------------------------------------------------------------------------
__global__ __launch_bounds__(256)
void cvt_f32_f16(const float* __restrict__ in, _Float16* __restrict__ out, int n) {
    int i = (blockIdx.x * blockDim.x + threadIdx.x) * 8;
    if (i >= n) return;
    f4 a = *(const f4*)(in + i);
    f4 b = *(const f4*)(in + i + 4);
    h8 o;
#pragma unroll
    for (int j = 0; j < 4; ++j) { o[j] = (_Float16)a[j]; o[j + 4] = (_Float16)b[j]; }
    *(h8*)(out + i) = o;
}

// ---------------------------------------------------------------------------
// Kernel 2/4: C[M,N] = A[M,K] * W[N,K]^T + bias   (f16 inputs, f32 accum)
// MODE 0: QKV projection -> n<2E written f16 row-major into qkv,
//         n>=2E (the V part) written TRANSPOSED into Vt[B*H*D][L] (packed b128)
// MODE 1: f32 output into Cf (final out-projection -> d_out)
// Block: 128 threads = 4 waves (2x2), 128x128 tile; wave tile 64x64 (16 WMMAs/k-step)
// ---------------------------------------------------------------------------
template <int MODE>
__global__ __launch_bounds__(128)
void gemm_wt(const _Float16* __restrict__ A, const _Float16* __restrict__ W,
             const float* __restrict__ bias,
             _Float16* __restrict__ Cq, _Float16* __restrict__ Vt,
             float* __restrict__ Cf,
             int M, int N, int K) {
    const int lane = threadIdx.x & 31;
    const int wave = threadIdx.x >> 5;
    const int l16  = lane & 15;
    const int hi   = lane >> 4;                 // 0 or 1
    const int bm   = blockIdx.y * 128 + (wave >> 1) * 64;
    const int bn   = blockIdx.x * 128 + (wave & 1) * 64;

    v8f acc[4][4];
#pragma unroll
    for (int i = 0; i < 4; ++i)
#pragma unroll
        for (int j = 0; j < 4; ++j) { v8f z = {}; acc[i][j] = z; }

    const int aOff = hi << 3;   // A-frag: elems 0..7 = k+{0..7|8..15}, 8..15 = +16
    const int bOff = hi << 3;   // B-frag: same k order as A-frag

    for (int k = 0; k < K; k += 32) {
        v16h af[4], bf[4];
#pragma unroll
        for (int t = 0; t < 4; ++t) {
            const _Float16* pa = A + (size_t)(bm + t * 16 + l16) * K + k + aOff;
            af[t] = load_frag(pa, pa + 16);
            const _Float16* pb = W + (size_t)(bn + t * 16 + l16) * K + k + bOff;
            bf[t] = load_frag(pb, pb + 16);
        }
#pragma unroll
        for (int tm = 0; tm < 4; ++tm)
#pragma unroll
            for (int tn = 0; tn < 4; ++tn)
                acc[tm][tn] = wmma16(af[tm], bf[tn], acc[tm][tn]);
    }

    // Epilogue. C layout: VGPR v -> row (v + hi*8), lanes l16 -> column.
#pragma unroll
    for (int tm = 0; tm < 4; ++tm) {
#pragma unroll
        for (int tn = 0; tn < 4; ++tn) {
            const int m0 = bm + tm * 16;
            const int n0 = bn + tn * 16;
            const int n  = n0 + l16;
            const float bb = bias[n];
            if (MODE == 1) {
#pragma unroll
                for (int v = 0; v < 8; ++v)
                    Cf[(size_t)(m0 + v + hi * 8) * N + n] = acc[tm][tn][v] + bb;
            } else {
                if (n0 < 2 * E_) {
#pragma unroll
                    for (int v = 0; v < 8; ++v)
                        Cq[(size_t)(m0 + v + hi * 8) * N + n] =
                            (_Float16)(acc[tm][tn][v] + bb);
                } else {
                    // V part -> transposed store: each lane's 8 accum values are
                    // 8 consecutive l positions -> one packed b128 store.
                    const int nv = n - 2 * E_;            // h*64 + d  in [0,1024)
                    const int b  = m0 >> 11;              // m / L
                    const int l0 = (m0 & (L_ - 1)) + hi * 8;
                    h8 pk;
#pragma unroll
                    for (int v = 0; v < 8; ++v)
                        pk[v] = (_Float16)(acc[tm][tn][v] + bb);
                    *(h8*)(Vt + (size_t)(b * (H_ * D_) + nv) * L_ + l0) = pk;
                }
            }
        }
    }
}

// ---------------------------------------------------------------------------
// Kernel 3: flash attention with additive knowledge bias.
// grid.x = B*H*(L/64); block = 128 threads = 4 waves; wave owns a 16-row q strip.
// Per 64-key block: S (8 WMMAs) -> *scale + fp32 bias -> online softmax
// (shfl_xor row reductions) -> P via LDS relayout -> O += P*V (8 WMMAs).
// ---------------------------------------------------------------------------
__global__ __launch_bounds__(128)
void attn_flash(const _Float16* __restrict__ qkv,  // [B*L][3E] (Q at 0, K at E)
                const _Float16* __restrict__ Vt,   // [B*H*D][L]
                const float* __restrict__ kbias,   // [B*H][L][L]
                _Float16* __restrict__ ctx) {      // [B*L][E]
    const int lane = threadIdx.x & 31;
    const int wave = threadIdx.x >> 5;
    const int l16  = lane & 15;
    const int hi   = lane >> 4;

    const int blk = blockIdx.x;
    const int qt  = blk & (L_ / 64 - 1);       // q tile
    const int bh  = blk >> 5;                  // b*H + h
    const int b   = bh >> 4;
    const int h   = bh & (H_ - 1);
    const int q0  = qt * 64 + wave * 16;       // this wave's q strip base

    const _Float16* Qb = qkv + (size_t)(b * L_ + q0 + l16) * EQ_ + h * D_;
    const _Float16* Kb = qkv + (size_t)(b * L_) * EQ_ + E_ + h * D_;
    const _Float16* Vb = Vt + (size_t)(bh * D_) * L_;
    const float*    Bb = kbias + (size_t)bh * L_ * L_;

    __shared__ _Float16 lds_p[4][16][64];      // wave-private P strips (8 KB)

    // Q A-fragments (fixed for whole kernel): d chunks 0..31 and 32..63
    const _Float16* qp = Qb + (hi << 3);
    v16h aq0 = load_frag(qp, qp + 16);
    v16h aq1 = load_frag(qp + 32, qp + 48);

    v8f o[4];
#pragma unroll
    for (int t = 0; t < 4; ++t) { v8f z = {}; o[t] = z; }
    float mrow[8], lrow[8];
#pragma unroll
    for (int v = 0; v < 8; ++v) { mrow[v] = -3.0e38f; lrow[v] = 0.f; }

    const float scale = 0.125f;                // 1/sqrt(64)

    for (int kb = 0; kb < L_; kb += 64) {
        // ---- S = Q*K^T (16 x 64 strip) ----
        v8f s[4];
#pragma unroll
        for (int t = 0; t < 4; ++t) {
            const _Float16* kp = Kb + (size_t)(kb + t * 16 + l16) * EQ_ + (hi << 3);
            v16h bk0 = load_frag(kp, kp + 16);
            v16h bk1 = load_frag(kp + 32, kp + 48);
            v8f z = {};
            s[t] = wmma16(aq0, bk0, z);
            s[t] = wmma16(aq1, bk1, s[t]);
        }
        // prefetch next bias block into GL2
        __builtin_prefetch(Bb + (size_t)(q0 + hi * 8) * L_ + kb + 64, 0, 1);

        // ---- scale + additive knowledge bias (the 537 MB fp32 stream) ----
#pragma unroll
        for (int t = 0; t < 4; ++t)
#pragma unroll
            for (int v = 0; v < 8; ++v)
                s[t][v] = s[t][v] * scale +
                          Bb[(size_t)(q0 + v + hi * 8) * L_ + kb + t * 16 + l16];

        // ---- online softmax: row max over 64 cols ----
        float mn[8];
#pragma unroll
        for (int v = 0; v < 8; ++v)
            mn[v] = fmaxf(fmaxf(s[0][v], s[1][v]), fmaxf(s[2][v], s[3][v]));
#pragma unroll
        for (int v = 0; v < 8; ++v) {
#pragma unroll
            for (int m = 1; m < 16; m <<= 1)
                mn[v] = fmaxf(mn[v], __shfl_xor(mn[v], m, 32));
            mn[v] = fmaxf(mn[v], mrow[v]);
        }
        float corr[8], rs[8];
#pragma unroll
        for (int v = 0; v < 8; ++v) { corr[v] = __expf(mrow[v] - mn[v]); rs[v] = 0.f; }

        // ---- P = exp(S - m), stash strip to LDS for C->A relayout ----
#pragma unroll
        for (int t = 0; t < 4; ++t)
#pragma unroll
            for (int v = 0; v < 8; ++v) {
                float p = __expf(s[t][v] - mn[v]);
                rs[v] += p;
                lds_p[wave][v + hi * 8][t * 16 + l16] = (_Float16)p;
            }
#pragma unroll
        for (int v = 0; v < 8; ++v) {
#pragma unroll
            for (int m = 1; m < 16; m <<= 1)
                rs[v] += __shfl_xor(rs[v], m, 32);
            lrow[v] = lrow[v] * corr[v] + rs[v];
            mrow[v] = mn[v];
        }
#pragma unroll
        for (int t = 0; t < 4; ++t)
#pragma unroll
            for (int v = 0; v < 8; ++v) o[t][v] *= corr[v];

        // same-wave LDS is in-order, but make the store->load dependency explicit
        asm volatile("s_wait_dscnt 0x0" ::: "memory");

        // ---- O += P * V ----
        const _Float16* pp = &lds_p[wave][l16][hi << 3];
        v16h ap0 = load_frag(pp, pp + 16);          // key cols 0..31
        v16h ap1 = load_frag(pp + 32, pp + 48);     // key cols 32..63
#pragma unroll
        for (int t = 0; t < 4; ++t) {
            const _Float16* vp = Vb + (size_t)(t * 16 + l16) * L_ + kb + (hi << 3);
            v16h bv0 = load_frag(vp, vp + 16);
            v16h bv1 = load_frag(vp + 32, vp + 48);
            o[t] = wmma16(ap0, bv0, o[t]);
            o[t] = wmma16(ap1, bv1, o[t]);
        }
    }

    // ---- normalize + write ctx (f16, [B*L][E] with E index = h*D + d) ----
    float inv[8];
#pragma unroll
    for (int v = 0; v < 8; ++v) inv[v] = 1.0f / lrow[v];
#pragma unroll
    for (int t = 0; t < 4; ++t)
#pragma unroll
        for (int v = 0; v < 8; ++v)
            ctx[(size_t)(b * L_ + q0 + v + hi * 8) * E_ + h * D_ + t * 16 + l16] =
                (_Float16)(o[t][v] * inv[v]);
}

// ---------------------------------------------------------------------------
// Host launcher
// ---------------------------------------------------------------------------
extern "C" void kernel_launch(void* const* d_in, const int* in_sizes, int n_in,
                              void* d_out, int out_size, void* d_ws, size_t ws_size,
                              hipStream_t stream) {
    const float* X       = (const float*)d_in[0];   // [B,L,E]
    const float* ke_bias = (const float*)d_in[1];   // [B*H,L,L]
    const float* Wqkv    = (const float*)d_in[2];   // [3E,E]
    const float* bqkv    = (const float*)d_in[3];   // [3E]
    const float* Wout    = (const float*)d_in[4];   // [E,E]
    const float* bout    = (const float*)d_in[5];   // [E]
    float* out = (float*)d_out;                     // [B,L,E]

    char* ws = (char*)d_ws;
    _Float16* Xh    = (_Float16*)(ws);                        //  8 MB
    _Float16* Wqkvh = (_Float16*)(ws + ((size_t)8  << 20));   //  6 MB
    _Float16* Wouth = (_Float16*)(ws + ((size_t)14 << 20));   //  2 MB
    _Float16* qkvh  = (_Float16*)(ws + ((size_t)16 << 20));   // 24 MB (Q,K used)
    _Float16* Vth   = (_Float16*)(ws + ((size_t)40 << 20));   //  8 MB [B*H*D][L]
    _Float16* ctxh  = (_Float16*)(ws + ((size_t)48 << 20));   //  8 MB

    const int M = B_ * L_;          // 4096

    // 1) fp32 -> fp16 conversions
    cvt_f32_f16<<<(B_ * L_ * E_) / (256 * 8), 256, 0, stream>>>(X, Xh, B_ * L_ * E_);
    cvt_f32_f16<<<(3 * E_ * E_) / (256 * 8), 256, 0, stream>>>(Wqkv, Wqkvh, 3 * E_ * E_);
    cvt_f32_f16<<<(E_ * E_) / (256 * 8), 256, 0, stream>>>(Wout, Wouth, E_ * E_);

    // 2) QKV projection (V written transposed)
    {
        dim3 grid(EQ_ / 128, M / 128);
        gemm_wt<0><<<grid, 128, 0, stream>>>(Xh, Wqkvh, bqkv, qkvh, Vth, nullptr,
                                             M, EQ_, E_);
    }

    // 3) flash attention with knowledge bias
    attn_flash<<<B_ * H_ * (L_ / 64), 128, 0, stream>>>(qkvh, Vth, ke_bias, ctxh);

    // 4) output projection (fp32 result)
    {
        dim3 grid(E_ / 128, M / 128);
        gemm_wt<1><<<grid, 128, 0, stream>>>(ctxh, Wouth, bout, nullptr, nullptr, out,
                                             M, E_, E_);
    }
    (void)in_sizes; (void)n_in; (void)out_size; (void)ws_size;
}
